// GraphUNet_53171695124574
// MI455X (gfx1250) — matrix operand order repaired and checked by reading, not compiled, not run on hardware
//
#include <hip/hip_runtime.h>
#include <math.h>

typedef __attribute__((ext_vector_type(16))) _Float16 v16h;
typedef __attribute__((ext_vector_type(8)))  float    v8f;

#define SENT_U 0x007FFFFFu  // fmap(-inf)

// ---------- monotonic float<->uint mapping (order-preserving) ----------
__device__ __forceinline__ unsigned fmap(float f) {
  unsigned u = __float_as_uint(f);
  return (u & 0x80000000u) ? ~u : (u | 0x80000000u);
}
__device__ __forceinline__ float funmap(unsigned u) {
  unsigned b = (u & 0x80000000u) ? (u & 0x7FFFFFFFu) : ~u;
  return __uint_as_float(b);
}

// ---------- small utility kernels ----------
__global__ void fill_u32(unsigned* p, unsigned v, int n) {
  int i = blockIdx.x * blockDim.x + threadIdx.x;
  if (i < n) p[i] = v;
}

__global__ void edge_cast(const long long* __restrict__ ei, int* __restrict__ s,
                          int* __restrict__ d, int E_) {
  int e = blockIdx.x * blockDim.x + threadIdx.x;
  if (e >= E_) return;
  s[e] = (int)ei[e];
  d[e] = (int)ei[(size_t)E_ + e];
}

// pack f32 weight (K x validCols, row-major) into WMMA B-fragment order f16.
// layout: P[(((ks*nT)+tile)*32 + lane)*16 + j] = W[(ks*32 + half*16 + j), tile*16+col]
__global__ void pack_b(const float* __restrict__ W, _Float16* __restrict__ P,
                       int K, int nT, int validCols) {
  int t = blockIdx.x * blockDim.x + threadIdx.x;
  int total = (K >> 5) * nT * 512;
  if (t >= total) return;
  int j    = t & 15;
  int lane = (t >> 4) & 31;
  int rest = t >> 9;
  int tile = rest % nT;
  int ks   = rest / nT;
  int kk = (ks << 5) + ((lane >> 4) << 4) + j;
  int c  = tile * 16 + (lane & 15);
  float v = (c < validCols) ? W[(size_t)kk * validCols + c] : 0.f;
  P[t] = (_Float16)v;
}

// ---------- scatter max (segment_max over edges) ----------
__global__ void scatter_max(const int* __restrict__ src, const int* __restrict__ dst,
                            const float* __restrict__ x, unsigned* __restrict__ agg,
                            int E_, int C) {
  int t = blockIdx.x * blockDim.x + threadIdx.x;
  int nc4 = C >> 2;
  int total = E_ * nc4;
  if (t >= total) return;
  int e = t / nc4;
  int q = (t - e * nc4) << 2;
  int s = src[e], d = dst[e];
  const float4 m = *(const float4*)(x + (size_t)s * C + q);
  unsigned* ap = agg + (size_t)d * C + q;
  atomicMax(ap + 0, fmap(m.x));
  atomicMax(ap + 1, fmap(m.y));
  atomicMax(ap + 2, fmap(m.z));
  atomicMax(ap + 3, fmap(m.w));
}

// ---------- pooling: scores ----------
__global__ void reduce_invnorm(const float* __restrict__ w, float* out) {
  __shared__ float sm[256];
  int t = threadIdx.x;
  float v = w[t];
  sm[t] = v * v;
  __syncthreads();
  for (int s = 128; s > 0; s >>= 1) {
    if (t < s) sm[t] += sm[t + s];
    __syncthreads();
  }
  if (t == 0) out[0] = rsqrtf(sm[0]);
}

__global__ void score_kernel(const float* __restrict__ x, const float* __restrict__ w,
                             const float* __restrict__ invn, float* __restrict__ val,
                             unsigned* __restrict__ key, int n) {
  int i = blockIdx.x * blockDim.x + threadIdx.x;
  if (i >= n) return;
  const float4* xr = (const float4*)(x + (size_t)i * 256);
  const float4* wr = (const float4*)w;
  float acc = 0.f;
#pragma unroll 4
  for (int j = 0; j < 64; ++j) {
    float4 a = xr[j], b = wr[j];
    acc += a.x * b.x + a.y * b.y + a.z * b.z + a.w * b.w;
  }
  float s = tanhf(acc * invn[0]);
  val[i] = s;
  key[i] = fmap(s);
}

// ---------- device-side radix top-k select ----------
__global__ void rs_init(unsigned* st, unsigned k) {
  if (threadIdx.x == 0 && blockIdx.x == 0) { st[0] = 0u; st[1] = k; }
}

__global__ void rs_hist(const unsigned* __restrict__ keys, int n,
                        const unsigned* __restrict__ st, unsigned* __restrict__ hist,
                        int shift) {
  int i = blockIdx.x * blockDim.x + threadIdx.x;
  if (i >= n) return;
  unsigned k = keys[i];
  unsigned mask = (shift >= 24) ? 0u : (0xFFFFFFFFu << (shift + 8));
  if ((k & mask) == st[0]) atomicAdd(&hist[(k >> shift) & 0xFFu], 1u);
}

__global__ void rs_scan(unsigned* st, const unsigned* __restrict__ hist, int shift) {
  if (threadIdx.x != 0 || blockIdx.x != 0) return;
  unsigned need = st[1];
  unsigned prefix = st[0];
  for (int b = 255; b >= 0; --b) {
    unsigned c = hist[b];
    if (c < need) { need -= c; }
    else { st[0] = prefix | ((unsigned)b << shift); st[1] = need; return; }
  }
  st[0] = prefix; st[1] = need;
}

__global__ void select_kernel(const unsigned* __restrict__ keys, const unsigned* __restrict__ st,
                              int n, int* __restrict__ mapping, int* __restrict__ perm,
                              unsigned* __restrict__ cnt) {
  int i = blockIdx.x * blockDim.x + threadIdx.x;
  if (i >= n) return;
  unsigned k = keys[i], thr = st[0], need = st[1];
  bool sel = false;
  if (k > thr) sel = true;
  else if (k == thr) {
    unsigned t = atomicAdd(&cnt[1], 1u);
    if (t < need) sel = true;
  }
  if (sel) {
    int p = (int)atomicAdd(&cnt[0], 1u);
    perm[p] = i;
    mapping[i] = p;
  }
}

__global__ void gather_scale(const float* __restrict__ xo, const float* __restrict__ val,
                             const int* __restrict__ perm, float* __restrict__ xn, int k) {
  int t = blockIdx.x * blockDim.x + threadIdx.x;
  int total = k * 256;
  if (t >= total) return;
  int j = t >> 8, c = t & 255;
  int o = perm[j];
  xn[t] = xo[(size_t)o * 256 + c] * val[o];
}

__global__ void relabel(const int* __restrict__ sp, const int* __restrict__ dp,
                        const int* __restrict__ mapping, int* __restrict__ sn,
                        int* __restrict__ dn, int E_, int k) {
  int e = blockIdx.x * blockDim.x + threadIdx.x;
  if (e >= E_) return;
  int ns = mapping[sp[e]];
  int nd = mapping[dp[e]];
  bool valid = (ns >= 0) && (nd >= 0);
  sn[e] = valid ? ns : 0;
  dn[e] = valid ? nd : k;
}

// res[perm[j]] += xs[j]  (perm entries unique -> no atomics)
__global__ void add_scatter(const float* __restrict__ xs, const int* __restrict__ perm,
                            float* __restrict__ res, int k) {
  int t = blockIdx.x * blockDim.x + threadIdx.x;
  int total = k * 256;
  if (t >= total) return;
  int j = t >> 8, c = t & 255;
  res[(size_t)perm[j] * 256 + c] += xs[t];
}

// ---------- fused SAGE GEMM: out = unmap(agg)@Wl + X@Wr + b, act ----------
__device__ __forceinline__ void store_tile(v8f acc, int tile, int row0, int lane, int M,
                                           int validCols, const float* __restrict__ bias,
                                           float* __restrict__ out, int act) {
  int n = lane & 15;
  int gc = tile * 16 + n;
  if (gc >= validCols) return;
  float bv = bias[gc];
  int mo = (lane >> 4) * 8;
#pragma unroll
  for (int v = 0; v < 8; ++v) {
    int gr = row0 + mo + v;
    if (gr < M) {
      float r = acc[v] + bv;
      if (act == 1) r = fmaxf(r, 0.f);
      else if (act == 2) r = tanhf(r) * 0.5f;
      out[(size_t)gr * validCols + gc] = r;
    }
  }
}

__global__ void __launch_bounds__(256)
sage_gemm(const unsigned* __restrict__ aggU, const float* __restrict__ X,
          const _Float16* __restrict__ Pl, const _Float16* __restrict__ Pr,
          const float* __restrict__ bias, float* __restrict__ out,
          int M, int K, int nT, int validCols, int act) {
  __shared__ __align__(32) _Float16 sA[512];  // agg tile in A-fragment order
  __shared__ __align__(32) _Float16 sX[512];  // x   tile in A-fragment order
  const int tid = threadIdx.x;
  const int wave = tid >> 5, lane = tid & 31;
  const int row0 = blockIdx.x * 16;
  const int t0 = wave, t1 = wave + 8;
  v8f acc0 = {};
  v8f acc1 = {};
  const int nK = K >> 5;
  for (int ks = 0; ks < nK; ++ks) {
    // cooperative load of 16x32 tiles, converting f32->f16, agg unmapped
    for (int e = tid; e < 512; e += 256) {
      int r = e >> 5, kk = e & 31;
      int gr = row0 + r;
      int gk = (ks << 5) + kk;
      // A-fragment position: half from K pattern {0..15}|{16..31}
      int half = kk >> 4;
      int dlane = half * 16 + r;
      int j = kk & 15;
      int di = dlane * 16 + j;
      float va = 0.f, vx = 0.f;
      if (gr < M) {
        float t = funmap(aggU[(size_t)gr * K + gk]);
        if (t > -1e37f && t < 1e37f) va = t;  // empty segments (-inf) -> 0
        vx = X[(size_t)gr * K + gk];
      }
      sA[di] = (_Float16)va;
      sX[di] = (_Float16)vx;
    }
    __syncthreads();
    v16h a  = *(const v16h*)(sA + lane * 16);
    v16h ax = *(const v16h*)(sX + lane * 16);
    if (t0 < nT) {
      v16h bl = *(const v16h*)(Pl + ((size_t)(ks * nT + t0) * 32 + lane) * 16);
      v16h br = *(const v16h*)(Pr + ((size_t)(ks * nT + t0) * 32 + lane) * 16);
      acc0 = __builtin_amdgcn_wmma_f32_16x16x32_f16(false, a,  false, bl, (short)0, acc0, false, false);
      acc0 = __builtin_amdgcn_wmma_f32_16x16x32_f16(false, ax, false, br, (short)0, acc0, false, false);
    }
    if (t1 < nT) {
      v16h bl = *(const v16h*)(Pl + ((size_t)(ks * nT + t1) * 32 + lane) * 16);
      v16h br = *(const v16h*)(Pr + ((size_t)(ks * nT + t1) * 32 + lane) * 16);
      acc1 = __builtin_amdgcn_wmma_f32_16x16x32_f16(false, a,  false, bl, (short)0, acc1, false, false);
      acc1 = __builtin_amdgcn_wmma_f32_16x16x32_f16(false, ax, false, br, (short)0, acc1, false, false);
    }
    __syncthreads();
  }
  if (t0 < nT) store_tile(acc0, t0, row0, lane, M, validCols, bias, out, act);
  if (t1 < nT) store_tile(acc1, t1, row0, lane, M, validCols, bias, out, act);
}

// =====================================================================
extern "C" void kernel_launch(void* const* d_in, const int* in_sizes, int n_in,
                              void* d_out, int out_size, void* d_ws, size_t ws_size,
                              hipStream_t stream) {
  (void)in_sizes; (void)n_in; (void)out_size; (void)ws_size;
  const int N0 = 50000, E = 800000, CIN = 32, CH = 256;
  const int K1 = 25000, K2 = 12500, K3 = 6250;

  const float*     x_in  = (const float*)d_in[0];
  const long long* ei    = (const long long*)d_in[1];
  const float*     w0l   = (const float*)d_in[2];
  const float*     w0r   = (const float*)d_in[3];
  const float*     b0    = (const float*)d_in[4];
  const float*     dwl   = (const float*)d_in[5];
  const float*     dwr   = (const float*)d_in[6];
  const float*     db    = (const float*)d_in[7];
  const float*     poolw = (const float*)d_in[8];
  const float*     uwl   = (const float*)d_in[9];
  const float*     uwr   = (const float*)d_in[10];
  const float*     ubv   = (const float*)d_in[11];
  const float*     uwlL  = (const float*)d_in[12];
  const float*     uwrL  = (const float*)d_in[13];
  const float*     ubL   = (const float*)d_in[14];

  char* base = (char*)d_ws;
  size_t off = 0;
  auto alloc = [&](size_t bytes) -> void* {
    void* p = base + off;
    off = (off + bytes + 255) & ~(size_t)255;
    return p;
  };
  auto cdiv = [](long long a, long long b) { return (unsigned)((a + b - 1) / b); };

  _Float16* Pw0l = (_Float16*)alloc((size_t)CIN * CH * 2);
  _Float16* Pw0r = (_Float16*)alloc((size_t)CIN * CH * 2);
  _Float16 *Pdl[3], *Pdr[3];
  for (int i = 0; i < 3; ++i) {
    Pdl[i] = (_Float16*)alloc((size_t)CH * CH * 2);
    Pdr[i] = (_Float16*)alloc((size_t)CH * CH * 2);
  }
  _Float16 *Pul[2], *Pur[2];
  for (int i = 0; i < 2; ++i) {
    Pul[i] = (_Float16*)alloc((size_t)CH * CH * 2);
    Pur[i] = (_Float16*)alloc((size_t)CH * CH * 2);
  }
  _Float16* PlastL = (_Float16*)alloc((size_t)CH * 16 * 2);
  _Float16* PlastR = (_Float16*)alloc((size_t)CH * 16 * 2);

  int* src0 = (int*)alloc((size_t)E * 4); int* dst0 = (int*)alloc((size_t)E * 4);
  int* src1 = (int*)alloc((size_t)E * 4); int* dst1 = (int*)alloc((size_t)E * 4);
  int* src2 = (int*)alloc((size_t)E * 4); int* dst2 = (int*)alloc((size_t)E * 4);
  int* src3 = (int*)alloc((size_t)E * 4); int* dst3 = (int*)alloc((size_t)E * 4);

  float* X0 = (float*)alloc((size_t)N0 * CH * 4);
  float* X1 = (float*)alloc((size_t)K1 * CH * 4);
  float* X2 = (float*)alloc((size_t)K2 * CH * 4);
  float* X3 = (float*)alloc((size_t)K3 * CH * 4);
  float* XP = (float*)alloc((size_t)K1 * CH * 4);  // pooled input scratch (max K1)
  float* U2 = (float*)alloc((size_t)K2 * CH * 4);
  float* U1 = (float*)alloc((size_t)K1 * CH * 4);

  unsigned* AGG = (unsigned*)alloc((size_t)(N0 + 1) * CH * 4);
  float*    VAL = (float*)alloc((size_t)N0 * 4);
  unsigned* KEY = (unsigned*)alloc((size_t)N0 * 4);
  int* PERM1 = (int*)alloc((size_t)K1 * 4);
  int* PERM2 = (int*)alloc((size_t)K2 * 4);
  int* PERM3 = (int*)alloc((size_t)K3 * 4);
  int* MAPPING = (int*)alloc((size_t)(N0 + 1) * 4);
  unsigned* HIST  = (unsigned*)alloc(256 * 4);
  unsigned* STATE = (unsigned*)alloc(16);
  unsigned* CNT   = (unsigned*)alloc(16);
  float*    INVN  = (float*)alloc(16);

  // ---- pack weights into B-fragment layout (f16) ----
  pack_b<<<cdiv((long long)CIN * CH, 256), 256, 0, stream>>>(w0l, Pw0l, CIN, 16, CH);
  pack_b<<<cdiv((long long)CIN * CH, 256), 256, 0, stream>>>(w0r, Pw0r, CIN, 16, CH);
  for (int i = 0; i < 3; ++i) {
    pack_b<<<cdiv((long long)CH * CH, 256), 256, 0, stream>>>(dwl + (size_t)i * CH * CH, Pdl[i], CH, 16, CH);
    pack_b<<<cdiv((long long)CH * CH, 256), 256, 0, stream>>>(dwr + (size_t)i * CH * CH, Pdr[i], CH, 16, CH);
  }
  for (int i = 0; i < 2; ++i) {
    pack_b<<<cdiv((long long)CH * CH, 256), 256, 0, stream>>>(uwl + (size_t)i * CH * CH, Pul[i], CH, 16, CH);
    pack_b<<<cdiv((long long)CH * CH, 256), 256, 0, stream>>>(uwr + (size_t)i * CH * CH, Pur[i], CH, 16, CH);
  }
  pack_b<<<cdiv((long long)CH * 16, 256), 256, 0, stream>>>(uwlL, PlastL, CH, 1, 3);
  pack_b<<<cdiv((long long)CH * 16, 256), 256, 0, stream>>>(uwrL, PlastR, CH, 1, 3);

  edge_cast<<<cdiv(E, 256), 256, 0, stream>>>(ei, src0, dst0, E);

  // ---- helpers ----
  auto conv = [&](const float* xin, int n, const int* s, const int* d,
                  const _Float16* Pl, const _Float16* Pr, const float* bias,
                  float* outp, int nT, int validCols, int act, int Kdim) {
    int aggN = (n + 1) * Kdim;
    fill_u32<<<cdiv(aggN, 256), 256, 0, stream>>>(AGG, SENT_U, aggN);
    scatter_max<<<cdiv((long long)E * (Kdim >> 2), 256), 256, 0, stream>>>(s, d, xin, AGG, E, Kdim);
    sage_gemm<<<cdiv(n, 16), 256, 0, stream>>>(AGG, xin, Pl, Pr, bias, outp, n, Kdim, nT, validCols, act);
  };

  auto pool = [&](const float* xin, int n, int k, const float* pw, int* perm,
                  const int* sp, const int* dp, int* sn, int* dn, float* xout) {
    reduce_invnorm<<<1, 256, 0, stream>>>(pw, INVN);
    score_kernel<<<cdiv(n, 128), 128, 0, stream>>>(xin, pw, INVN, VAL, KEY, n);
    rs_init<<<1, 1, 0, stream>>>(STATE, (unsigned)k);
    for (int shift = 24; shift >= 0; shift -= 8) {
      (void)hipMemsetAsync(HIST, 0, 256 * sizeof(unsigned), stream);
      rs_hist<<<cdiv(n, 256), 256, 0, stream>>>(KEY, n, STATE, HIST, shift);
      rs_scan<<<1, 1, 0, stream>>>(STATE, HIST, shift);
    }
    (void)hipMemsetAsync(CNT, 0, 2 * sizeof(unsigned), stream);
    (void)hipMemsetAsync(MAPPING, 0xFF, (size_t)(n + 1) * sizeof(int), stream);
    select_kernel<<<cdiv(n, 256), 256, 0, stream>>>(KEY, STATE, n, MAPPING, perm, CNT);
    gather_scale<<<cdiv((long long)k * CH, 256), 256, 0, stream>>>(xin, VAL, perm, xout, k);
    relabel<<<cdiv(E, 256), 256, 0, stream>>>(sp, dp, MAPPING, sn, dn, E, k);
  };

  // ---- down path ----
  conv(x_in, N0, src0, dst0, Pw0l, Pw0r, b0, X0, 16, CH, 1, CIN);

  pool(X0, N0, K1, poolw + 0 * CH, PERM1, src0, dst0, src1, dst1, XP);
  conv(XP, K1, src1, dst1, Pdl[0], Pdr[0], db + 0 * CH, X1, 16, CH, 1, CH);

  pool(X1, K1, K2, poolw + 1 * CH, PERM2, src1, dst1, src2, dst2, XP);
  conv(XP, K2, src2, dst2, Pdl[1], Pdr[1], db + 1 * CH, X2, 16, CH, 1, CH);

  pool(X2, K2, K3, poolw + 2 * CH, PERM3, src2, dst2, src3, dst3, XP);
  conv(XP, K3, src3, dst3, Pdl[2], Pdr[2], db + 2 * CH, X3, 16, CH, 1, CH);

  // ---- up path ----
  add_scatter<<<cdiv((long long)K3 * CH, 256), 256, 0, stream>>>(X3, PERM3, X2, K3);
  conv(X2, K2, src2, dst2, Pul[0], Pur[0], ubv + 0 * CH, U2, 16, CH, 1, CH);

  add_scatter<<<cdiv((long long)K2 * CH, 256), 256, 0, stream>>>(U2, PERM2, X1, K2);
  conv(X1, K1, src1, dst1, Pul[1], Pur[1], ubv + 1 * CH, U1, 16, CH, 1, CH);

  add_scatter<<<cdiv((long long)K1 * CH, 256), 256, 0, stream>>>(U1, PERM1, X0, K1);
  // final conv: 3 output cols padded to one 16-col tile; act=2 fuses tanh(x)*0.5
  conv(X0, N0, src0, dst0, PlastL, PlastR, ubL, (float*)d_out, 1, 3, 2, CH);
}
